// TaggingFNNCRFDecoder_1760936592043
// MI455X (gfx1250) — matrix operand order repaired and checked by reading, not compiled, not run on hardware
//
#include <hip/hip_runtime.h>

// Problem sizes (fixed by the reference)
#define BB   256          // batch
#define TT   512          // seq len
#define HH   1024         // hidden
#define KTAG 128          // num tags
#define BT   (BB * TT)    // 131072 rows for the GEMM

typedef __attribute__((ext_vector_type(16))) __bf16 v16bf;
typedef __attribute__((ext_vector_type(8)))  float  v8f;
typedef __attribute__((ext_vector_type(4)))  float  f4v;

// ---------------------------------------------------------------------------
// fp32 -> (bf16_hi, bf16_lo) split: x ~= hi + lo; after the 3-term WMMA
// product expansion (ah*bh + ah*bl + al*bh) total accuracy is ~fp32.
// Non-temporal loads: hiddens is streamed once (512 MB), keep it out of L2
// so W_hi/W_lo and logits stay resident.
// ---------------------------------------------------------------------------
__device__ __forceinline__ void cvt8_nt(const float* __restrict__ p,
                                        v16bf& hi, v16bf& lo, int base) {
  f4v u = __builtin_nontemporal_load((const f4v*)p);
  f4v v = __builtin_nontemporal_load((const f4v*)(p + 4));
  float f[8] = {u.x, u.y, u.z, u.w, v.x, v.y, v.z, v.w};
#pragma unroll
  for (int i = 0; i < 8; ++i) {
    __bf16 h = (__bf16)f[i];
    hi[base + i] = h;
    lo[base + i] = (__bf16)(f[i] - (float)h);
  }
}

__device__ __forceinline__ v8f wmma_bf16(v16bf a, v16bf b, v8f c) {
  // D = A(16x32 bf16) x B(32x16 bf16) + C(16x16 f32)
  return __builtin_amdgcn_wmma_f32_16x16x32_bf16(
      false, a, false, b, (short)0, c, false, false);
}

// ---------------------------------------------------------------------------
// Kernel 0: one-shot split of W[128,1024] into bf16 hi/lo (256 KB each,
// L2-resident). Removes all B-side conversion ALU from the GEMM hot loop.
// ---------------------------------------------------------------------------
__global__ __launch_bounds__(256) void convert_w_kernel(
    const float* __restrict__ W, __bf16* __restrict__ whi,
    __bf16* __restrict__ wlo) {
  const int idx = blockIdx.x * 256 + threadIdx.x;   // grid covers 128*1024
  const float x = W[idx];
  const __bf16 h = (__bf16)x;
  whi[idx] = h;
  wlo[idx] = (__bf16)(x - (float)h);
}

// ---------------------------------------------------------------------------
// Kernel 1: logits[BT,128] = hiddens[BT,1024] @ W[128,1024]^T + b
// One wave32 owns a 32(M) x 128(N) slab: hiddens touched exactly once from
// HBM (~512MB, NT), W_hi/W_lo re-read from L2. 8 waves / block.
// A fragment (16x32 bf16): lane r=lane&15 holds row r; kg=lane>>4 selects
//   K = kg*16 + 0..15 contiguous (elems 0-15), matching B.
// B fragment (32x16 bf16): lane holds W row (n-tile*16 + r), K = kg*16+0..15
//   contiguous (elems 0-15) -> direct v16bf load.
// C/D: elem i -> row = base + i + 8*kg, col = ntile*16 + r.
// ---------------------------------------------------------------------------
__global__ __launch_bounds__(256) void gemm_logits_kernel(
    const float* __restrict__ hiddens, const __bf16* __restrict__ whi,
    const __bf16* __restrict__ wlo, const float* __restrict__ bias,
    float* __restrict__ logits) {
  const int lane = threadIdx.x & 31;
  const int wave = threadIdx.x >> 5;
  const int rowBase = (blockIdx.x * 8 + wave) * 32;
  const int r  = lane & 15;
  const int kg = lane >> 4;

  v8f acc[2][8];
#pragma unroll
  for (int m = 0; m < 2; ++m)
#pragma unroll
    for (int nt = 0; nt < 8; ++nt) acc[m][nt] = (v8f)0.0f;

  const float* a0p = hiddens + (size_t)(rowBase + r) * HH;
  const float* a1p = hiddens + (size_t)(rowBase + 16 + r) * HH;

  for (int kt = 0; kt < HH / 32; ++kt) {
    const int ka = kt * 32 + kg * 16;   // A chunk base
    const int kb = kt * 32 + kg * 16;   // B chunk base
    v16bf a0h, a0l, a1h, a1l;
    cvt8_nt(a0p + ka,      a0h, a0l, 0);
    cvt8_nt(a0p + ka + 8,  a0h, a0l, 8);
    cvt8_nt(a1p + ka,      a1h, a1l, 0);
    cvt8_nt(a1p + ka + 8,  a1h, a1l, 8);
#pragma unroll
    for (int nt = 0; nt < 8; ++nt) {
      const size_t boff = (size_t)(nt * 16 + r) * HH + kb;
      const v16bf bh = *(const v16bf*)(whi + boff);
      const v16bf bl = *(const v16bf*)(wlo + boff);
      // 3-term split product; interleave the two acc chains for the scheduler
      acc[0][nt] = wmma_bf16(a0h, bh, acc[0][nt]);
      acc[1][nt] = wmma_bf16(a1h, bh, acc[1][nt]);
      acc[0][nt] = wmma_bf16(a0h, bl, acc[0][nt]);
      acc[1][nt] = wmma_bf16(a1h, bl, acc[1][nt]);
      acc[0][nt] = wmma_bf16(a0l, bh, acc[0][nt]);
      acc[1][nt] = wmma_bf16(a1l, bh, acc[1][nt]);
    }
  }

#pragma unroll
  for (int nt = 0; nt < 8; ++nt) {
    const int col = nt * 16 + r;
    const float bv = bias[col];
#pragma unroll
    for (int i = 0; i < 8; ++i) {
      const int row0 = rowBase + i + 8 * kg;
      logits[(size_t)row0 * KTAG + col]        = acc[0][nt][i] + bv;
      logits[(size_t)(row0 + 16) * KTAG + col] = acc[1][nt][i] + bv;
    }
  }
}

// ---------------------------------------------------------------------------
// Kernel 2: Viterbi decode. One block (=batch), 128 threads (one per tag).
// LDS (129 KB, CDNA5 320KB/WGP): transitions (64KB) + ping-pong scores +
// full backpointer history (64KB) so backtracking never leaves the WGP.
// ---------------------------------------------------------------------------
__global__ __launch_bounds__(128) void viterbi_kernel(
    const float* __restrict__ logits, const float* __restrict__ trans,
    const float* __restrict__ startv, const float* __restrict__ endv,
    const int* __restrict__ mask, float* __restrict__ pred) {
  extern __shared__ char smem[];
  float* s_trans = (float*)smem;                        // 128*128
  float* s_a     = (float*)(smem + KTAG * KTAG * 4);    // 128
  float* s_b     = s_a + KTAG;                          // 128
  unsigned char* s_hist = (unsigned char*)(s_b + KTAG); // (T-1)*128

  const int b = blockIdx.x;
  const int k = threadIdx.x;
  for (int i = k; i < KTAG * KTAG; i += 128) s_trans[i] = trans[i];

  const float* lg = logits + (size_t)b * TT * KTAG;
  s_a[k] = startv[k] + lg[k];
  __syncthreads();

  float* cur = s_a;
  float* nxt = s_b;
  for (int t = 1; t < TT; ++t) {
    float best = -3.402823466e38f;
    int   bj   = 0;
#pragma unroll 4
    for (int j = 0; j < KTAG; ++j) {
      float v = cur[j] + s_trans[j * KTAG + k];
      if (v > best) { best = v; bj = j; }    // strict '>' -> lowest-index tie, matches argmax
    }
    const bool m = (mask[(size_t)b * TT + t] != 0);
    const float e = lg[(size_t)t * KTAG + k];
    nxt[k] = m ? (best + e) : cur[k];
    s_hist[(size_t)(t - 1) * KTAG + k] = (unsigned char)(m ? bj : k);
    __syncthreads();
    float* tmp = cur; cur = nxt; nxt = tmp;
  }

  nxt[k] = cur[k] + endv[k];
  __syncthreads();
  if (k == 0) {
    float best = nxt[0]; int tag = 0;
    for (int j = 1; j < KTAG; ++j)
      if (nxt[j] > best) { best = nxt[j]; tag = j; }
    float* pb = pred + (size_t)b * TT;
    pb[TT - 1] = (float)tag;
    for (int t = TT - 2; t >= 0; --t) {      // LDS-resident pointer chase
      tag = s_hist[(size_t)t * KTAG + tag];
      pb[t] = (float)tag;
    }
  }
}

// ---------------------------------------------------------------------------
// Kernel 3: CRF NLL per batch: (den - num). Forward log-sum-exp DP with
// LDS-resident transitions; gold-path score reduced across the block.
// ---------------------------------------------------------------------------
__global__ __launch_bounds__(128) void crf_nll_kernel(
    const float* __restrict__ logits, const float* __restrict__ trans,
    const float* __restrict__ startv, const float* __restrict__ endv,
    const int* __restrict__ mask, const int* __restrict__ labels,
    float* __restrict__ batch_loss) {
  extern __shared__ char smem[];
  float* s_trans = (float*)smem;                     // 128*128
  float* s_a     = (float*)(smem + KTAG * KTAG * 4); // 128
  float* s_b     = s_a + KTAG;                       // 128
  float* s_red   = s_b + KTAG;                       // 128
  int*   s_cnt   = (int*)(s_red + KTAG);             // 128

  const int b = blockIdx.x;
  const int k = threadIdx.x;
  for (int i = k; i < KTAG * KTAG; i += 128) s_trans[i] = trans[i];

  const float* lg = logits + (size_t)b * TT * KTAG;
  s_a[k] = startv[k] + lg[k];
  __syncthreads();

  float* cur = s_a;
  float* nxt = s_b;
  for (int t = 1; t < TT; ++t) {
    float mx = -3.402823466e38f;
#pragma unroll 4
    for (int j = 0; j < KTAG; ++j)
      mx = fmaxf(mx, cur[j] + s_trans[j * KTAG + k]);
    float s = 0.0f;
#pragma unroll 4
    for (int j = 0; j < KTAG; ++j)
      s += __expf(cur[j] + s_trans[j * KTAG + k] - mx);
    const bool m = (mask[(size_t)b * TT + t] != 0);
    const float e = lg[(size_t)t * KTAG + k];
    nxt[k] = m ? (mx + __logf(s) + e) : cur[k];
    __syncthreads();
    float* tmp = cur; cur = nxt; nxt = tmp;
  }

  nxt[k] = cur[k] + endv[k];   // final scores + end transitions
  // gold-path score, strided over t across the block
  const int* lab = labels + (size_t)b * TT;
  const int* mk  = mask   + (size_t)b * TT;
  float part = 0.0f; int cnt = 0;
  for (int t = k; t < TT; t += 128) {
    const int   l  = lab[t];
    const int   mi = (mk[t] != 0);
    const float mm = (float)mi;
    part += lg[(size_t)t * KTAG + l] * mm;
    if (t >= 1) part += s_trans[lab[t - 1] * KTAG + l] * mm;
    cnt += mi;
  }
  s_red[k] = part; s_cnt[k] = cnt;
  __syncthreads();

  if (k == 0) {
    float mx = -3.402823466e38f;
    for (int j = 0; j < KTAG; ++j) mx = fmaxf(mx, nxt[j]);
    float s = 0.0f;
    for (int j = 0; j < KTAG; ++j) s += __expf(nxt[j] - mx);
    const float den = mx + __logf(s);
    float num = startv[lab[0]];
    int total = 0;
    for (int j = 0; j < 128; ++j) { num += s_red[j]; total += s_cnt[j]; }
    num += endv[lab[total - 1]];
    batch_loss[b] = den - num;
  }
}

__global__ void loss_reduce_kernel(const float* __restrict__ bl,
                                   float* __restrict__ out) {
  __shared__ float red[BB];
  const int k = threadIdx.x;
  red[k] = bl[k];
  __syncthreads();
  for (int s = BB / 2; s > 0; s >>= 1) {
    if (k < s) red[k] += red[k + s];
    __syncthreads();
  }
  if (k == 0) out[0] = red[0] / (float)BB;
}

// ---------------------------------------------------------------------------
extern "C" void kernel_launch(void* const* d_in, const int* in_sizes, int n_in,
                              void* d_out, int out_size, void* d_ws, size_t ws_size,
                              hipStream_t stream) {
  (void)in_sizes; (void)n_in; (void)out_size; (void)ws_size;
  const float* hiddens = (const float*)d_in[0];
  const int*   mask    = (const int*)d_in[1];
  const int*   labels  = (const int*)d_in[2];
  const float* W       = (const float*)d_in[3];
  const float* bias    = (const float*)d_in[4];
  const float* startv  = (const float*)d_in[5];
  const float* endv    = (const float*)d_in[6];
  const float* trans   = (const float*)d_in[7];

  float* out    = (float*)d_out;
  // Workspace layout:
  //   [0, BT*KTAG)            logits  (64 MiB f32)
  //   [BT*KTAG, +BB)          per-batch losses
  //   then W_hi, W_lo         (128*1024 bf16 each)
  float*  logits = (float*)d_ws;
  float*  bl     = logits + (size_t)BT * KTAG;
  __bf16* whi    = (__bf16*)(bl + BB);
  __bf16* wlo    = whi + (size_t)KTAG * HH;

  // 0) Split W into bf16 hi/lo once (512 blocks x 256 threads covers 128K).
  convert_w_kernel<<<(KTAG * HH) / 256, 256, 0, stream>>>(W, whi, wlo);

  // 1) WMMA GEMM: 512 blocks x 8 waves, 32 rows/wave.
  gemm_logits_kernel<<<BT / 256, 256, 0, stream>>>(hiddens, whi, wlo, bias, logits);

  // 2) Viterbi: 129 KB LDS/block (needs CDNA5 320KB WGP LDS).
  const size_t vit_lds = (size_t)KTAG * KTAG * 4 + 2 * KTAG * 4 + (size_t)(TT - 1) * KTAG;
  (void)hipFuncSetAttribute((const void*)viterbi_kernel,
                            hipFuncAttributeMaxDynamicSharedMemorySize, (int)vit_lds);
  viterbi_kernel<<<BB, KTAG, vit_lds, stream>>>(logits, trans, startv, endv, mask, out);

  // 3) NLL forward DP: 66 KB LDS/block.
  const size_t nll_lds = (size_t)KTAG * KTAG * 4 + 4 * KTAG * 4;
  (void)hipFuncSetAttribute((const void*)crf_nll_kernel,
                            hipFuncAttributeMaxDynamicSharedMemorySize, (int)nll_lds);
  crf_nll_kernel<<<BB, KTAG, nll_lds, stream>>>(logits, trans, startv, endv, mask, labels, bl);

  // 4) mean over batch -> d_out[B*T]
  loss_reduce_kernel<<<1, BB, 0, stream>>>(bl, out + (size_t)BT);
}
